// Attention_12266426598027
// MI455X (gfx1250) — matrix operand order repaired and checked by reading, not compiled, not run on hardware
//
#include <hip/hip_runtime.h>
#include <hip/hip_bf16.h>

// ---------------------------------------------------------------------------
// Problem constants
// ---------------------------------------------------------------------------
constexpr int B_   = 4;
constexpr int S_   = 2048;
constexpr int H_   = 896;     // hidden
constexpr int NH_  = 14;      // query heads
constexpr int NKV_ = 2;       // kv heads
constexpr int HD_  = 64;      // head dim
constexpr int NREP = NH_ / NKV_;          // 7
constexpr int M_TOK = B_ * S_;            // 8192 tokens
// q pre-scale: (1/sqrt(64)) * log2(e), so softmax uses exp2(s - m)
#define QSCALE 0.18033688f

typedef _Float16 v16h __attribute__((ext_vector_type(16)));
typedef _Float16 v8h  __attribute__((ext_vector_type(8)));
typedef float    v8f  __attribute__((ext_vector_type(8)));
typedef unsigned int v4u __attribute__((ext_vector_type(4)));
typedef unsigned int v8u __attribute__((ext_vector_type(8)));

union HV { v16h v; v8h h[2]; };

// ---------------------------------------------------------------------------
// fp32 -> f16 pack (vectorized x4); n must be divisible by 4*256
// ---------------------------------------------------------------------------
__global__ __launch_bounds__(256) void cvt_f16(const float* __restrict__ src,
                                               _Float16* __restrict__ dst)
{
  int i = (blockIdx.x * 256 + threadIdx.x) * 4;
  float4 d = *reinterpret_cast<const float4*>(src + i);
  _Float16* p = dst + i;
  p[0] = (_Float16)d.x; p[1] = (_Float16)d.y;
  p[2] = (_Float16)d.z; p[3] = (_Float16)d.w;
}

// ---------------------------------------------------------------------------
// TDM tile load: copy a (128 rows x 64 f16) tile from a row-major f16 tensor
// (row stride = tensor_cols elements) into LDS at byte offset lds_off, with
// 4 DWORDs of padding inserted after every 32 DWORDs (one 64-half row), so
// the LDS row stride is 72 halfs (144 B, 16B-aligned).
//
// D# per CDNA5 ISA:
//  group0: [1:0]=count=1, [63:32]=lds_addr, [120:64]=global_addr, [127:126]=2
//  group1: [17:16]=data_size(1=2B), [20]=pad_en, [24:22]=pad_interval(4=32dw),
//          [31:25]=pad_amount(3=4dw), [79:48]=tensor_dim0, [111:80]=tensor_dim1,
//          [127:112]=tile_dim0(64), [143:128]=tile_dim1(128),
//          [207:160]=tensor_dim0_stride
//  groups2/3: zero (tile_dim2..4 = 0 => unused; 2-D transfer)
// Issued once per wave (EXEC ignored); caller gates to a single wave and uses
// S_WAIT_TENSORCNT + workgroup barrier to publish the tile.
// ---------------------------------------------------------------------------
__device__ static inline void tdm_load_tile_f16(const _Float16* gsrc,
                                                unsigned lds_off,
                                                unsigned tensor_cols)
{
  unsigned long long ga = (unsigned long long)(const void*)gsrc;
  v4u g0;
  g0[0] = 1u;                                    // count=1, user mode
  g0[1] = lds_off;                               // LDS byte address
  g0[2] = (unsigned)ga;                          // global_addr[31:0]
  g0[3] = ((unsigned)(ga >> 32) & 0x01FFFFFFu) | (2u << 30); // [56:32] | type=2

  v8u g1;
  g1[0] = (1u << 16)        // data_size = 2 bytes
        | (1u << 20)        // pad_enable
        | (4u << 22)        // pad_interval: 32 DWORDs
        | (3u << 25);       // pad_amount: 4 DWORDs
  g1[1] = (tensor_cols & 0xFFFFu) << 16;                    // tensor_dim0 lo
  g1[2] = (tensor_cols >> 16) | (0u << 16);                 // dim0 hi | dim1 lo
  g1[3] = (1u << 4) /* tensor_dim1 hi16 of (1<<20) */ | (64u << 16); // tile_dim0=64
  g1[4] = 128u;                                             // tile_dim1 = 128 rows
  g1[5] = tensor_cols;                                      // dim0_stride lo32
  g1[6] = 0u;
  g1[7] = 0u;

  v4u g2 = {0u, 0u, 0u, 0u};
  v4u g3 = {0u, 0u, 0u, 0u};

  asm volatile("tensor_load_to_lds %0, %1, %2, %3"
               :: "s"(g0), "s"(g1), "s"(g2), "s"(g3)
               : "memory");
}

// ---------------------------------------------------------------------------
// WMMA GEMM: C[M,N] = A[M,K] @ W[N,K]^T (+ bias[N]); A and W are f16, C fp32.
// Block tile 128x128, BK=64, 256 threads = 8 waves (4 row x 2 col); each wave
// owns a 32x64 sub-tile = 2x4 grid of 16x16 fp32 accumulators, 16 WMMA/iter.
//
// Double-buffered TDM pipeline: wave 0 issues the DMA for tile i+1 into the
// alternate LDS buffer, then waits TENSORcnt<=2 (TDM completes in-order, so
// <=2 outstanding means tile i's two transfers are done while tile i+1's are
// still in flight), barriers, and all waves run WMMAs on tile i while the
// Tensor Data Mover fills the other buffer. The trailing barrier guarantees
// buffer i is fully consumed before tile i+2 (issued after it) overwrites it.
// Requires M%128==0, N%128==0, K%64==0 (true for all launches here).
// ---------------------------------------------------------------------------
template <bool HAS_BIAS>
__global__ __launch_bounds__(256) void gemm_wmma(
    const _Float16* __restrict__ A, const _Float16* __restrict__ W,
    const float* __restrict__ bias, float* __restrict__ C,
    int M, int N, int K)
{
  __shared__ __align__(16) _Float16 As[2][128 * 72]; // row stride 72 halfs
  __shared__ __align__(16) _Float16 Bs[2][128 * 72];

  const int tid  = threadIdx.x;
  const int lane = tid & 31;
  const int wave = tid >> 5;
  const int wr   = wave & 3;   // wave row (0..3) -> 32-row slab
  const int wc   = wave >> 2;  // wave col (0..1) -> 64-col slab
  const int ln   = lane & 15;
  const int g    = lane >> 4;

  const int m0 = blockIdx.x * 128;
  const int n0 = blockIdx.y * 128;

  // LDS byte offsets of the staging tiles (generic ptr low 32 bits = LDS addr)
  unsigned offA[2], offB[2];
  offA[0] = (unsigned)(unsigned long long)(void*)&As[0][0];
  offA[1] = (unsigned)(unsigned long long)(void*)&As[1][0];
  offB[0] = (unsigned)(unsigned long long)(void*)&Bs[0][0];
  offB[1] = (unsigned)(unsigned long long)(void*)&Bs[1][0];

  v8f acc[2][4];
  const v8f zero8 = {0.f,0.f,0.f,0.f,0.f,0.f,0.f,0.f};
#pragma unroll
  for (int ms = 0; ms < 2; ++ms)
#pragma unroll
    for (int ns = 0; ns < 4; ++ns) acc[ms][ns] = zero8;

  const int KT = K >> 6;   // number of 64-wide K tiles

  // prologue: stage tile 0 into buffer 0
  if (wave == 0) {
    tdm_load_tile_f16(A + (size_t)m0 * K, offA[0], (unsigned)K);
    tdm_load_tile_f16(W + (size_t)n0 * K, offB[0], (unsigned)K);
  }

  for (int it = 0; it < KT; ++it) {
    const int buf = it & 1;
    if (wave == 0) {
      if (it + 1 < KT) {
        const int kk = (it + 1) << 6;
        tdm_load_tile_f16(A + (size_t)m0 * K + kk, offA[buf ^ 1], (unsigned)K);
        tdm_load_tile_f16(W + (size_t)n0 * K + kk, offB[buf ^ 1], (unsigned)K);
        __builtin_amdgcn_s_wait_tensorcnt(2);  // tile `it` complete (in-order)
      } else {
        __builtin_amdgcn_s_wait_tensorcnt(0);
      }
    }
    __syncthreads();

    const _Float16* Asb = &As[buf][0];
    const _Float16* Bsb = &Bs[buf][0];

    // A 16x32 f16 frag: lane holds row m=ln; K chunks [8g,8g+8),[16+8g,16+8g+8)
    HV afrag[2][2];
#pragma unroll
    for (int ms = 0; ms < 2; ++ms) {
      int mrow = wr * 32 + ms * 16 + ln;
#pragma unroll
      for (int c = 0; c < 2; ++c) {
        afrag[ms][c].h[0] =
            *reinterpret_cast<const v8h*>(&Asb[mrow * 72 + c * 32 + 8 * g]);
        afrag[ms][c].h[1] =
            *reinterpret_cast<const v8h*>(&Asb[mrow * 72 + c * 32 + 16 + 8 * g]);
      }
    }
    // B 32x16 f16 frag: lane holds column n=ln; K chunks [8g,8g+8),[16+8g,16+8g+8)
    HV bfrag[4][2];
#pragma unroll
    for (int ns = 0; ns < 4; ++ns) {
      int nrow = wc * 64 + ns * 16 + ln;
#pragma unroll
      for (int c = 0; c < 2; ++c) {
        bfrag[ns][c].h[0] =
            *reinterpret_cast<const v8h*>(&Bsb[nrow * 72 + c * 32 + 8 * g]);
        bfrag[ns][c].h[1] =
            *reinterpret_cast<const v8h*>(&Bsb[nrow * 72 + c * 32 + 16 + 8 * g]);
      }
    }
#pragma unroll
    for (int ms = 0; ms < 2; ++ms)
#pragma unroll
      for (int ns = 0; ns < 4; ++ns) {
        acc[ms][ns] = __builtin_amdgcn_wmma_f32_16x16x32_f16(
            false, afrag[ms][0].v, false, bfrag[ns][0].v,
            (short)0, acc[ms][ns], false, false);
        acc[ms][ns] = __builtin_amdgcn_wmma_f32_16x16x32_f16(
            false, afrag[ms][1].v, false, bfrag[ns][1].v,
            (short)0, acc[ms][ns], false, false);
      }
    __syncthreads();
  }

  // store: C layout lane -> (m = r + 8g, n = ln)
#pragma unroll
  for (int ms = 0; ms < 2; ++ms)
#pragma unroll
    for (int ns = 0; ns < 4; ++ns) {
      int col = n0 + wc * 64 + ns * 16 + ln;
      float bv = HAS_BIAS ? bias[col] : 0.0f;
#pragma unroll
      for (int r = 0; r < 8; ++r) {
        int row = m0 + wr * 32 + ms * 16 + r + 8 * g;
        C[(size_t)row * N + col] = acc[ms][ns][r] + bv;
      }
    }
}

// ---------------------------------------------------------------------------
// RoPE + pack to f16: Q [b,h,s,d] (pre-scaled), K [b,kv,s,d], V^T [b,kv,d,s]
// ---------------------------------------------------------------------------
__global__ __launch_bounds__(128) void rope_pack(
    const float* __restrict__ qf, const float* __restrict__ kf,
    const float* __restrict__ vf, const int* __restrict__ pos_ids,
    _Float16* __restrict__ Qo, _Float16* __restrict__ Ko,
    _Float16* __restrict__ Vt)
{
  int idx   = blockIdx.x * 128 + threadIdx.x;   // exact: B*S*896 / 128 blocks
  int c     = idx % H_;
  int token = idx / H_;
  int s = token % S_;
  int b = token / S_;
  int d = c & 63;
  int h = c >> 6;

  float pos = (float)pos_ids[s];
  int   i   = d & 31;
  // inv_freq = theta^{-(2i/64)} = exp(-(i/32) * ln(1e6))
  float inv = __expf(-(float)i * 0.43173384f);
  float ang = pos * inv;
  float cs = __cosf(ang);
  float sn = __sinf(ang);

  int pd = (d < 32) ? d + 32 : d - 32;

  float x  = qf[(size_t)token * H_ + c];
  float xo = qf[(size_t)token * H_ + h * 64 + pd];
  float rq = (d < 32) ? (x * cs - xo * sn) : (x * cs + xo * sn);
  Qo[((size_t)(b * NH_ + h) * S_ + s) * HD_ + d] = (_Float16)(rq * QSCALE);

  if (c < NKV_ * HD_) {
    int kvh = h;  // c < 128 -> h in {0,1}
    float kx = kf[(size_t)token * (NKV_ * HD_) + c];
    float ko = kf[(size_t)token * (NKV_ * HD_) + kvh * 64 + pd];
    float rk = (d < 32) ? (kx * cs - ko * sn) : (kx * cs + ko * sn);
    Ko[((size_t)(b * NKV_ + kvh) * S_ + s) * HD_ + d] = (_Float16)rk;

    float vx = vf[(size_t)token * (NKV_ * HD_) + c];
    Vt[((size_t)(b * NKV_ + kvh) * HD_ + d) * S_ + s] = (_Float16)vx;
  }
}

// ---------------------------------------------------------------------------
// Causal flash attention with WMMA, GQA (kvh = h / 7).
// grid = (S/64, B*NH); 128 threads = 4 waves; wave w handles query rows
// [bx*64 + 16w, +16). Keys in 64-wide blocks, bx+1 blocks per wave (uniform),
// causal mask in the last block only. Online softmax with exp2.
// ---------------------------------------------------------------------------
__global__ __launch_bounds__(128) void flash_attn(
    const _Float16* __restrict__ Q, const _Float16* __restrict__ Kf,
    const _Float16* __restrict__ VT, _Float16* __restrict__ O)
{
  __shared__ __align__(16) _Float16 Pbuf[4][16][72]; // wave-private P tiles

  const int tid  = threadIdx.x;
  const int lane = tid & 31;
  const int wave = tid >> 5;
  const int ln   = lane & 15;
  const int g    = lane >> 4;

  const int bh  = blockIdx.y;
  const int b   = bh / NH_;
  const int h   = bh % NH_;
  const int kvh = h / NREP;
  const int q0  = blockIdx.x * 64 + wave * 16;

  const _Float16* qbase = Q  + ((size_t)(b * NH_ + h) * S_ + q0) * HD_;
  const _Float16* kbase = Kf + (size_t)(b * NKV_ + kvh) * S_ * HD_;
  const _Float16* vbase = VT + (size_t)(b * NKV_ + kvh) * HD_ * S_;

  // Q as A-fragments over d: chunks [0,32) and [32,64)
  HV qfrag[2];
#pragma unroll
  for (int c = 0; c < 2; ++c) {
    qfrag[c].h[0] = *reinterpret_cast<const v8h*>(qbase + ln * HD_ + c * 32 + 8 * g);
    qfrag[c].h[1] = *reinterpret_cast<const v8h*>(qbase + ln * HD_ + c * 32 + 16 + 8 * g);
  }

  float mrow[8], lrow[8];
  v8f accO[4];
  const v8f zero8 = {0.f,0.f,0.f,0.f,0.f,0.f,0.f,0.f};
#pragma unroll
  for (int r = 0; r < 8; ++r) { mrow[r] = -3.0e38f; lrow[r] = 0.0f; }
#pragma unroll
  for (int t = 0; t < 4; ++t) accO[t] = zero8;

  const int nblocks = blockIdx.x + 1;
  for (int blk = 0; blk < nblocks; ++blk) {
    const int kb = blk * 64;

    // ---- S = Q * K^T for a 16x64 score tile ----
    v8f accS[4];
#pragma unroll
    for (int t = 0; t < 4; ++t) accS[t] = zero8;
#pragma unroll
    for (int t = 0; t < 4; ++t) {
      int key = kb + t * 16 + ln;            // B-frag: lane holds column = key
#pragma unroll
      for (int c = 0; c < 2; ++c) {          // contraction chunks d=[32c,32c+32)
        HV bf;
        bf.h[0] = *reinterpret_cast<const v8h*>(kbase + (size_t)key * HD_ + c * 32 + 8 * g);
        bf.h[1] = *reinterpret_cast<const v8h*>(kbase + (size_t)key * HD_ + c * 32 + 16 + 8 * g);
        accS[t] = __builtin_amdgcn_wmma_f32_16x16x32_f16(
            false, qfrag[c].v, false, bf.v, (short)0, accS[t], false, false);
      }
    }

    // ---- causal mask (last block only) ----
    if (blk == nblocks - 1) {
#pragma unroll
      for (int t = 0; t < 4; ++t) {
        int key = kb + t * 16 + ln;
#pragma unroll
        for (int r = 0; r < 8; ++r) {
          int qr = q0 + r + 8 * g;
          if (key > qr) accS[t][r] = -3.0e38f;
        }
      }
    }

    // ---- online softmax: row max over the 64-key block ----
    float bmax[8];
#pragma unroll
    for (int r = 0; r < 8; ++r) {
      float m = accS[0][r];
#pragma unroll
      for (int t = 1; t < 4; ++t) m = fmaxf(m, accS[t][r]);
#pragma unroll
      for (int off = 1; off <= 8; off <<= 1)
        m = fmaxf(m, __shfl_xor(m, off));
      bmax[r] = m;
    }
    float alpha[8];
#pragma unroll
    for (int r = 0; r < 8; ++r) {
      float nm = fmaxf(mrow[r], bmax[r]);
      alpha[r] = exp2f(mrow[r] - nm);
      mrow[r]  = nm;
    }

    // ---- P = exp2(S - m), write to LDS in A-layout source form ----
    float bsum[8];
#pragma unroll
    for (int r = 0; r < 8; ++r) bsum[r] = 0.0f;
#pragma unroll
    for (int t = 0; t < 4; ++t)
#pragma unroll
      for (int r = 0; r < 8; ++r) {
        float p = exp2f(accS[t][r] - mrow[r]);
        bsum[r] += p;
        Pbuf[wave][r + 8 * g][t * 16 + ln] = (_Float16)p;
      }
#pragma unroll
    for (int r = 0; r < 8; ++r) {
      float s = bsum[r];
#pragma unroll
      for (int off = 1; off <= 8; off <<= 1) s += __shfl_xor(s, off);
      lrow[r] = lrow[r] * alpha[r] + s;
    }
#pragma unroll
    for (int t = 0; t < 4; ++t)
#pragma unroll
      for (int r = 0; r < 8; ++r) accO[t][r] *= alpha[r];

    // wave-private LDS RAW: DS ops are in-order within a wave; drain DScnt and
    // stop the compiler reordering before the fragment reads.
    asm volatile("s_wait_dscnt 0" ::: "memory");

    // ---- P as A-fragments (16 x 64 split into two K=32 chunks) ----
    HV pfrag[2];
#pragma unroll
    for (int c = 0; c < 2; ++c) {
      pfrag[c].h[0] = *reinterpret_cast<const v8h*>(&Pbuf[wave][ln][c * 32 + 8 * g]);
      pfrag[c].h[1] = *reinterpret_cast<const v8h*>(&Pbuf[wave][ln][c * 32 + 16 + 8 * g]);
    }

    // ---- O += P * V  (V^T layout: lane holds column d, contiguous keys) ----
#pragma unroll
    for (int t = 0; t < 4; ++t) {
      int d = t * 16 + ln;
#pragma unroll
      for (int c = 0; c < 2; ++c) {
        HV vf;
        vf.h[0] = *reinterpret_cast<const v8h*>(vbase + (size_t)d * S_ + kb + c * 32 + 8 * g);
        vf.h[1] = *reinterpret_cast<const v8h*>(vbase + (size_t)d * S_ + kb + c * 32 + 16 + 8 * g);
        accO[t] = __builtin_amdgcn_wmma_f32_16x16x32_f16(
            false, pfrag[c].v, false, vf.v, (short)0, accO[t], false, false);
      }
    }
  }

  // ---- epilogue: normalize, store attn in [b, s, h*64+d] (GEMM-A layout) ----
#pragma unroll
  for (int r = 0; r < 8; ++r) {
    float inv = 1.0f / lrow[r];
    int row = q0 + r + 8 * g;
#pragma unroll
    for (int t = 0; t < 4; ++t) {
      float val = accO[t][r] * inv;
      O[((size_t)(b * S_ + row) * NH_ + h) * HD_ + t * 16 + ln] = (_Float16)val;
    }
  }
}

// ---------------------------------------------------------------------------
// Launcher
// ---------------------------------------------------------------------------
extern "C" void kernel_launch(void* const* d_in, const int* in_sizes, int n_in,
                              void* d_out, int out_size, void* d_ws, size_t ws_size,
                              hipStream_t stream)
{
  const float* hidden = (const float*)d_in[0];
  const int*   posids = (const int*)  d_in[1];
  const float* Wq = (const float*)d_in[2];
  const float* bq = (const float*)d_in[3];
  const float* Wk = (const float*)d_in[4];
  const float* bk = (const float*)d_in[5];
  const float* Wv = (const float*)d_in[6];
  const float* bv = (const float*)d_in[7];
  const float* Wo = (const float*)d_in[8];
  float* out = (float*)d_out;

  // workspace carve-out (256B aligned)
  char* ws = (char*)d_ws;
  size_t off = 0;
  auto carve = [&](size_t bytes) {
    size_t o = off;
    off += (bytes + 255) & ~(size_t)255;
    return o;
  };
  float*    qf32 = (float*)   (ws + carve((size_t)M_TOK * H_ * 4));
  float*    kf32 = (float*)   (ws + carve((size_t)M_TOK * NKV_ * HD_ * 4));
  float*    vf32 = (float*)   (ws + carve((size_t)M_TOK * NKV_ * HD_ * 4));
  _Float16* Q16  = (_Float16*)(ws + carve((size_t)M_TOK * H_ * 2));
  _Float16* K16  = (_Float16*)(ws + carve((size_t)M_TOK * NKV_ * HD_ * 2));
  _Float16* VT16 = (_Float16*)(ws + carve((size_t)M_TOK * NKV_ * HD_ * 2));
  _Float16* A16  = (_Float16*)(ws + carve((size_t)M_TOK * H_ * 2));
  _Float16* X16  = (_Float16*)(ws + carve((size_t)M_TOK * H_ * 2));       // hidden f16
  _Float16* Wq16 = (_Float16*)(ws + carve((size_t)H_ * H_ * 2));
  _Float16* Wk16 = (_Float16*)(ws + carve((size_t)NKV_ * HD_ * H_ * 2));
  _Float16* Wv16 = (_Float16*)(ws + carve((size_t)NKV_ * HD_ * H_ * 2));
  _Float16* Wo16 = (_Float16*)(ws + carve((size_t)H_ * H_ * 2));

  // 0) pack fp32 -> f16 (all counts divisible by 1024)
  cvt_f16<<<(M_TOK * H_) / 1024, 256, 0, stream>>>(hidden, X16);
  cvt_f16<<<(H_ * H_) / 1024, 256, 0, stream>>>(Wq, Wq16);
  cvt_f16<<<(NKV_ * HD_ * H_) / 1024, 256, 0, stream>>>(Wk, Wk16);
  cvt_f16<<<(NKV_ * HD_ * H_) / 1024, 256, 0, stream>>>(Wv, Wv16);
  cvt_f16<<<(H_ * H_) / 1024, 256, 0, stream>>>(Wo, Wo16);

  dim3 blk(256);

  // 1) QKV projections (WMMA + double-buffered TDM staging, fp32 out, +bias)
  gemm_wmma<true><<<dim3(M_TOK / 128, H_ / 128), blk, 0, stream>>>(
      X16, Wq16, bq, qf32, M_TOK, H_, H_);
  gemm_wmma<true><<<dim3(M_TOK / 128, 1), blk, 0, stream>>>(
      X16, Wk16, bk, kf32, M_TOK, NKV_ * HD_, H_);
  gemm_wmma<true><<<dim3(M_TOK / 128, 1), blk, 0, stream>>>(
      X16, Wv16, bv, vf32, M_TOK, NKV_ * HD_, H_);

  // 2) RoPE + pack to f16 (Q pre-scaled; V transposed)
  rope_pack<<<(M_TOK * H_) / 128, 128, 0, stream>>>(
      qf32, kf32, vf32, posids, Q16, K16, VT16);

  // 3) causal flash attention (WMMA)
  flash_attn<<<dim3(S_ / 64, B_ * NH_), 128, 0, stream>>>(Q16, K16, VT16, A16);

  // 4) output projection -> d_out (fp32)
  gemm_wmma<false><<<dim3(M_TOK / 128, H_ / 128), blk, 0, stream>>>(
      A16, Wo16, nullptr, out, M_TOK, H_, H_);
}
